// RvT_64570538328183
// MI455X (gfx1250) — compile-verified, matrix operand order checked by reading, not run on hardware
//
#include <hip/hip_runtime.h>
#include <hip/hip_bf16.h>

// ---------------------------------------------------------------------------
// CDNA5 (gfx1250) implementation. Everything GEMM-shaped runs through
// v_wmma_f32_16x16x32_f16 (wave32). Exploits h0=c0=0 (f-gate & Wx_h dead).
// Conv weight slabs staged to LDS via GLOBAL_LOAD_ASYNC_TO_LDS_B128.
// ---------------------------------------------------------------------------

typedef __attribute__((ext_vector_type(16))) _Float16 v16h;
typedef __attribute__((ext_vector_type(8)))  float    v8f;

union FragH { v16h h; uint4 u[2]; };
union Pack8 { _Float16 h[8]; uint4 u; };

#define NB   8
#define CC   256
#define GG   8
#define HH   36
#define WW   36
#define HWSZ (HH*WW)          // 1296
#define DD   1024             // 32*32 valid-conv spatial

__device__ __forceinline__ v8f wmma_f16(v16h a, v16h b, v8f c) {
  return __builtin_amdgcn_wmma_f32_16x16x32_f16(false, a, false, b, (short)0, c,
                                                false, false);
}

// Load one 16x16x32 fragment row/col: 32 consecutive K-halves at `base`,
// lane-half selects K offset (kb = (lane>>4)*8): halves 0-7 <- K[kb..kb+7],
// halves 8-15 <- K[kb+16..kb+23].  Two 16B loads.
__device__ __forceinline__ v16h load_frag(const _Float16* base, int kb) {
  FragH f;
  f.u[0] = *reinterpret_cast<const uint4*>(base + kb);
  f.u[1] = *reinterpret_cast<const uint4*>(base + kb + 16);
  return f.h;
}

// ---------------------------------------------------------------- tau reduce
__global__ void k_tau(const float* __restrict__ tau, float* __restrict__ scale) {
  __shared__ float sm[256];
  int t = threadIdx.x;
  sm[t] = tau[t];
  __syncthreads();
  for (int s = 128; s > 0; s >>= 1) { if (t < s) sm[t] += sm[t + s]; __syncthreads(); }
  if (t == 0) *scale = sm[0];
}

// ------------------------------------------------- inp NCHW f32 -> [n][p][ci] f16
__global__ void k_inpT(const float* __restrict__ inp, _Float16* __restrict__ inpT) {
  int idx = blockIdx.x * 256 + threadIdx.x;            // N*128*1296 total
  if (idx >= NB * 128 * HWSZ) return;
  int p  = idx % HWSZ;
  int ci = (idx / HWSZ) % 128;
  int n  = idx / (HWSZ * 128);
  inpT[((size_t)n * HWSZ + p) * 128 + ci] = (_Float16)inp[idx];
}

// ---------------------------------------------------------- flat f32 -> f16
__global__ void k_cvt(const float* __restrict__ src, _Float16* __restrict__ dst, int n) {
  int idx = blockIdx.x * 256 + threadIdx.x;
  if (idx < n) dst[idx] = (_Float16)src[idx];
}

// ---- 5x5 grouped weights (C,64,5,5) f32 -> [g][tap][co32][ci32] f16 (x-half only)
__global__ void k_w5(const float* __restrict__ W, _Float16* __restrict__ wt) {
  int idx = blockIdx.x * 256 + threadIdx.x;            // 8*25*32*32 = 204800
  if (idx >= 204800) return;
  int ci  = idx & 31;
  int co  = (idx >> 5) & 31;
  int tap = (idx / 1024) % 25;
  int g   = idx / (1024 * 25);
  wt[idx] = (_Float16)W[(((size_t)(g * 32 + co) * 64) + ci) * 25 + tap];
}

// ----------------------------------------- x = Wx_in @ inp   (1x1 dense proj)
// out: channels-last f16 [n][1296][256]
__global__ void __launch_bounds__(256)
k_xproj(const _Float16* __restrict__ inpT, const _Float16* __restrict__ wx,
        _Float16* __restrict__ xout) {
  int lane = threadIdx.x & 31;
  int wv   = threadIdx.x >> 5;
  int cot  = blockIdx.y * 8 + wv;          // 16 co-tiles of 16
  int n    = blockIdx.z;
  int row  = lane & 15, hi = lane >> 4, kb = hi * 8;
  int pos  = blockIdx.x * 16 + row;        // 81 tiles, exact
  const _Float16* arow = wx + (size_t)(cot * 16 + row) * 128;
  const _Float16* brow = inpT + ((size_t)n * HWSZ + pos) * 128;
  v8f acc = {};
#pragma unroll
  for (int ko = 0; ko < 128; ko += 32)
    acc = wmma_f16(load_frag(arow + ko, kb), load_frag(brow + ko, kb), acc);
  Pack8 pk;
#pragma unroll
  for (int r = 0; r < 8; r++) pk.h[r] = (_Float16)acc[r];
  *reinterpret_cast<uint4*>(xout + ((size_t)n * HWSZ + pos) * CC + cot * 16 + 8 * hi) = pk.u;
}

// ------------------------------------------------- grouped 5x5 conv (implicit GEMM)
// mode 0: qT f16 [(n*8+g)*1296 + pos][32]   (padded, 36x36)
// mode 1: kT f16 [(n*8+g)*1024 + pos][32]   (valid, 32x32)
// mode 2: v  f16 [(n*8+g)*32 + c][1024]     (valid, 32x32)
// mode 3: pre f32 NCHW [n][256][1296]       (padded, 36x36)
__global__ void __launch_bounds__(256)
k_conv5(const _Float16* __restrict__ x, const _Float16* __restrict__ wt,
        void* __restrict__ outp, int pad, int Ho, int Wo, int tilesX, int mode) {
  __shared__ _Float16 s_w[25 * 32 * 32];   // 51200 B
  __shared__ _Float16 s_x[8 * 20 * 32];    // 10240 B
  int g = blockIdx.y, n = blockIdx.z;
  int ty = blockIdx.x / tilesX, tx = blockIdx.x % tilesX;
  int t = threadIdx.x;

  { // stage group weight slab (25600 halves = 3200 x b128) via async DMA to LDS
    const uint4* src = reinterpret_cast<const uint4*>(wt + (size_t)g * 25600);
    unsigned ldsbase = (unsigned)(uintptr_t)(void*)s_w;   // LDS aperture: low 32 bits
    for (int i = t; i < 3200; i += 256) {
      unsigned dst = ldsbase + (unsigned)i * 16u;
      const uint4* gsrc = src + i;
      asm volatile("global_load_async_to_lds_b128 %0, %1, off"
                   :: "v"(dst), "v"(gsrc) : "memory");
    }
  }
  { // stage 20x8 halo tile of this group's 32 channels (zero-pad OOB, predicated)
    int oy0 = ty * 4 - pad, ox0 = tx * 16 - pad;
    for (int i = t; i < 640; i += 256) {
      int pix = i >> 2, part = i & 3;
      int gy = oy0 + pix / 20, gx = ox0 + pix % 20;
      uint4 val = make_uint4(0u, 0u, 0u, 0u);
      if ((unsigned)gy < 36u && (unsigned)gx < 36u)
        val = *reinterpret_cast<const uint4*>(
            x + ((size_t)n * HWSZ + gy * WW + gx) * CC + g * 32 + part * 8);
      *reinterpret_cast<uint4*>(s_x + pix * 32 + part * 8) = val;
    }
  }
  asm volatile("s_wait_asynccnt 0" ::: "memory");
  __syncthreads();

  int lane = t & 31, wv = t >> 5;
  int cot = wv >> 2, pt = wv & 3;          // 2 co-tiles x 4 pos-rows
  int row = lane & 15, hi = lane >> 4, kb = hi * 8;
  v8f acc = {};
#pragma unroll
  for (int kh = 0; kh < 5; kh++)
#pragma unroll
    for (int kw = 0; kw < 5; kw++) {
      int tap = kh * 5 + kw;
      v16h a = load_frag(s_w + (size_t)(tap * 32 + cot * 16 + row) * 32, kb);
      v16h b = load_frag(s_x + (size_t)((pt + kh) * 20 + row + kw) * 32, kb);
      acc = wmma_f16(a, b, acc);
    }

  int oy = ty * 4 + pt, ox = tx * 16 + row;
  int pos = oy * Wo + ox;
  int HoWo = Ho * Wo;
  int mb = cot * 16 + 8 * hi;
  if (mode <= 1) {
    if (ox < Wo) {
      Pack8 pk;
#pragma unroll
      for (int r = 0; r < 8; r++) pk.h[r] = (_Float16)acc[r];
      _Float16* dst = (_Float16*)outp;
      *reinterpret_cast<uint4*>(dst + ((size_t)(n * GG + g) * HoWo + pos) * 32 + mb) = pk.u;
    }
  } else if (mode == 2) {
    _Float16* dst = (_Float16*)outp;
#pragma unroll
    for (int r = 0; r < 8; r++)
      dst[((size_t)(n * GG + g) * 32 + mb + r) * DD + pos] = (_Float16)acc[r];
  } else {
    if (ox < Wo) {
      float* dst = (float*)outp;
#pragma unroll
      for (int r = 0; r < 8; r++)
        dst[((size_t)n * CC + g * 32 + mb + r) * HWSZ + pos] = acc[r];
    }
  }
}

// ------------------------------------------------------ flash attention, 1 wave/block
// logits[q,d] = scale * sum_c q[c,q] k[c,d]; softmax over d; a[c,q] = sum_d v[c,d] p[q,d]
__global__ void __launch_bounds__(32)
k_flash(const _Float16* __restrict__ qT, const _Float16* __restrict__ kT,
        const _Float16* __restrict__ vv, const float* __restrict__ scalep,
        _Float16* __restrict__ aT) {
  __shared__ _Float16 p_lds[16 * 32];
  __shared__ float stat[16];
  int lane = threadIdx.x;
  int qt = blockIdx.x, h = blockIdx.y, n = blockIdx.z;
  int row = lane & 15, hi = lane >> 4, kb = hi * 8;
  float scale = *scalep;

  size_t nh = (size_t)(n * GG + h);
  v16h qf = load_frag(qT + (nh * HWSZ + qt * 16 + row) * 32, kb);
  const _Float16* kbase = kT + nh * DD * 32;
  const _Float16* vbase = vv + nh * 32 * DD;

  float rmax[8], rsum[8];
#pragma unroll
  for (int r = 0; r < 8; r++) { rmax[r] = -1e30f; rsum[r] = 0.f; }
  v8f acc0 = {}, acc1 = {};

  for (int dt = 0; dt < DD; dt += 32) {
    v16h b0 = load_frag(kbase + (size_t)(dt + row) * 32, kb);
    v16h b1 = load_frag(kbase + (size_t)(dt + 16 + row) * 32, kb);
    v8f z = {};
    v8f s0 = wmma_f16(qf, b0, z);
    v8f s1 = wmma_f16(qf, b1, z);
#pragma unroll
    for (int r = 0; r < 8; r++) {
      float a0 = s0[r] * scale, a1 = s1[r] * scale;
      float m = fmaxf(a0, a1);
      m = fmaxf(m, __shfl_xor(m, 1)); m = fmaxf(m, __shfl_xor(m, 2));
      m = fmaxf(m, __shfl_xor(m, 4)); m = fmaxf(m, __shfl_xor(m, 8));
      float nm = fmaxf(rmax[r], m);
      float cr = __expf(rmax[r] - nm);
      rmax[r] = nm;
      float p0 = __expf(a0 - nm), p1 = __expf(a1 - nm);
      float rs = p0 + p1;
      rs += __shfl_xor(rs, 1); rs += __shfl_xor(rs, 2);
      rs += __shfl_xor(rs, 4); rs += __shfl_xor(rs, 8);
      rsum[r] = rsum[r] * cr + rs;
      int m2 = r + 8 * hi;
      p_lds[m2 * 32 + row]      = (_Float16)p0;
      p_lds[m2 * 32 + 16 + row] = (_Float16)p1;
      if (row == 0) stat[m2] = cr;   // broadcast corr per q-row
    }
    __syncthreads();
    float cq = stat[row];            // column q of AV accumulators
#pragma unroll
    for (int r = 0; r < 8; r++) { acc0[r] *= cq; acc1[r] *= cq; }
    v16h pb  = load_frag(p_lds + (size_t)row * 32, kb);
    v16h va0 = load_frag(vbase + (size_t)row * DD + dt, kb);
    v16h va1 = load_frag(vbase + (size_t)(16 + row) * DD + dt, kb);
    acc0 = wmma_f16(va0, pb, acc0);
    acc1 = wmma_f16(va1, pb, acc1);
    __syncthreads();                 // p_lds / stat reuse next iteration
  }
  if (row == 0) {
#pragma unroll
    for (int r = 0; r < 8; r++) stat[r + 8 * hi] = rsum[r];
  }
  __syncthreads();
  float inv = 1.0f / stat[row];
  size_t abase = (nh * HWSZ + qt * 16 + row) * 32;
  Pack8 pk;
#pragma unroll
  for (int r = 0; r < 8; r++) pk.h[r] = (_Float16)(acc0[r] * inv);
  *reinterpret_cast<uint4*>(aT + abase + 8 * hi) = pk.u;
#pragma unroll
  for (int r = 0; r < 8; r++) pk.h[r] = (_Float16)(acc1[r] * inv);
  *reinterpret_cast<uint4*>(aT + abase + 16 + 8 * hi) = pk.u;
}

// ------------------------ gates: act = act_fn(gain*(Wa@a + pre_conv) + bias)
// 6 waves: gate(0=i sigmoid, 1=g tanh, 2=o sigmoid) x 2 co-tiles
__global__ void __launch_bounds__(192)
k_combine(const _Float16* __restrict__ aT,
          const _Float16* waI, const _Float16* waG, const _Float16* waO,
          float* preI, float* preG, float* preO,
          const float* gainI, const float* biasI,
          const float* gainG, const float* biasG,
          const float* gainO, const float* biasO) {
  int lane = threadIdx.x & 31, wv = threadIdx.x >> 5;
  int gate = wv >> 1, cot = wv & 1;
  int qt = blockIdx.x, g = blockIdx.y, n = blockIdx.z;
  int row = lane & 15, hi = lane >> 4, kb = hi * 8;

  const _Float16* wa = (gate == 0) ? waI : (gate == 1) ? waG : waO;
  float* pre         = (gate == 0) ? preI : (gate == 1) ? preG : preO;
  const float* gain  = (gate == 0) ? gainI : (gate == 1) ? gainG : gainO;
  const float* bias  = (gate == 0) ? biasI : (gate == 1) ? biasG : biasO;

  int pos = qt * 16 + row;
  v16h a = load_frag(wa + (size_t)(g * 32 + cot * 16 + row) * 32, kb);
  v16h b = load_frag(aT + ((size_t)(n * GG + g) * HWSZ + pos) * 32, kb);
  v8f z = {};
  v8f acc = wmma_f16(a, b, z);
#pragma unroll
  for (int r = 0; r < 8; r++) {
    int co = g * 32 + cot * 16 + 8 * hi + r;
    size_t idx = ((size_t)n * CC + co) * HWSZ + pos;
    float vl = acc[r] + pre[idx];
    vl = gain[co] * vl + bias[co];
    vl = (gate == 1) ? tanhf(vl) : (1.0f / (1.0f + __expf(-vl)));
    pre[idx] = vl;
  }
}

// --------------------------------------- h = o * tanh(i*g)   (c0 == 0)
__global__ void k_final(const float* __restrict__ ai, const float* __restrict__ ag,
                        const float* __restrict__ ao, float* __restrict__ out) {
  size_t idx = (size_t)blockIdx.x * 256 + threadIdx.x;
  if (idx >= (size_t)NB * CC * HWSZ) return;
  out[idx] = ao[idx] * tanhf(ai[idx] * ag[idx]);
}

// ===========================================================================
extern "C" void kernel_launch(void* const* d_in, const int* in_sizes, int n_in,
                              void* d_out, int out_size, void* d_ws, size_t ws_size,
                              hipStream_t stream) {
  (void)in_sizes; (void)n_in; (void)out_size; (void)ws_size;
  const float* inp    = (const float*)d_in[0];
  const float* Wx_in  = (const float*)d_in[2];
  const float* Wq     = (const float*)d_in[3];
  const float* Wk     = (const float*)d_in[4];
  const float* Wv     = (const float*)d_in[5];
  const float* tau    = (const float*)d_in[6];
  const float* Wa_i   = (const float*)d_in[7];
  const float* Wxh_i  = (const float*)d_in[8];
  const float* gain_i = (const float*)d_in[9];
  const float* bias_i = (const float*)d_in[10];
  const float* Wa_g   = (const float*)d_in[15];
  const float* Wxh_g  = (const float*)d_in[16];
  const float* gain_g = (const float*)d_in[17];
  const float* bias_g = (const float*)d_in[18];
  const float* Wa_o   = (const float*)d_in[19];
  const float* Wxh_o  = (const float*)d_in[20];
  const float* gain_o = (const float*)d_in[21];
  const float* bias_o = (const float*)d_in[22];

  char* w = (char*)d_ws;
  size_t off = 0;
  auto alloc = [&](size_t bytes) -> char* {
    char* p = w + off; off += (bytes + 255) & ~(size_t)255; return p;
  };
  float*     wsScale = (float*)alloc(256);
  _Float16*  inpT = (_Float16*)alloc((size_t)NB * HWSZ * 128 * 2);
  _Float16*  xf16 = (_Float16*)alloc((size_t)NB * HWSZ * CC * 2);
  _Float16*  wq5  = (_Float16*)alloc(204800 * 2);
  _Float16*  wk5  = (_Float16*)alloc(204800 * 2);
  _Float16*  wv5  = (_Float16*)alloc(204800 * 2);
  _Float16*  wi5  = (_Float16*)alloc(204800 * 2);
  _Float16*  wg5  = (_Float16*)alloc(204800 * 2);
  _Float16*  wo5  = (_Float16*)alloc(204800 * 2);
  _Float16*  wx1  = (_Float16*)alloc(CC * 128 * 2);
  _Float16*  waI  = (_Float16*)alloc(CC * 32 * 2);
  _Float16*  waG  = (_Float16*)alloc(CC * 32 * 2);
  _Float16*  waO  = (_Float16*)alloc(CC * 32 * 2);
  _Float16*  qT   = (_Float16*)alloc((size_t)NB * GG * HWSZ * 32 * 2);
  _Float16*  kTb  = (_Float16*)alloc((size_t)NB * GG * DD * 32 * 2);
  _Float16*  vvb  = (_Float16*)alloc((size_t)NB * GG * 32 * DD * 2);
  _Float16*  aT   = (_Float16*)alloc((size_t)NB * GG * HWSZ * 32 * 2);
  float*     preI = (float*)alloc((size_t)NB * CC * HWSZ * 4);
  float*     preG = (float*)alloc((size_t)NB * CC * HWSZ * 4);
  float*     preO = (float*)alloc((size_t)NB * CC * HWSZ * 4);

  k_tau<<<1, 256, 0, stream>>>(tau, wsScale);
  k_inpT<<<(NB * 128 * HWSZ) / 256, 256, 0, stream>>>(inp, inpT);
  k_cvt<<<128, 256, 0, stream>>>(Wx_in, wx1, CC * 128);
  k_cvt<<<32, 256, 0, stream>>>(Wa_i, waI, CC * 32);
  k_cvt<<<32, 256, 0, stream>>>(Wa_g, waG, CC * 32);
  k_cvt<<<32, 256, 0, stream>>>(Wa_o, waO, CC * 32);
  k_w5<<<800, 256, 0, stream>>>(Wq,    wq5);
  k_w5<<<800, 256, 0, stream>>>(Wk,    wk5);
  k_w5<<<800, 256, 0, stream>>>(Wv,    wv5);
  k_w5<<<800, 256, 0, stream>>>(Wxh_i, wi5);
  k_w5<<<800, 256, 0, stream>>>(Wxh_g, wg5);
  k_w5<<<800, 256, 0, stream>>>(Wxh_o, wo5);

  k_xproj<<<dim3(81, 2, NB), 256, 0, stream>>>(inpT, wx1, xf16);

  k_conv5<<<dim3(27, GG, NB), 256, 0, stream>>>(xf16, wq5, qT,  2, 36, 36, 3, 0);
  k_conv5<<<dim3(16, GG, NB), 256, 0, stream>>>(xf16, wk5, kTb, 0, 32, 32, 2, 1);
  k_conv5<<<dim3(16, GG, NB), 256, 0, stream>>>(xf16, wv5, vvb, 0, 32, 32, 2, 2);
  k_conv5<<<dim3(27, GG, NB), 256, 0, stream>>>(xf16, wi5, preI, 2, 36, 36, 3, 3);
  k_conv5<<<dim3(27, GG, NB), 256, 0, stream>>>(xf16, wg5, preG, 2, 36, 36, 3, 3);
  k_conv5<<<dim3(27, GG, NB), 256, 0, stream>>>(xf16, wo5, preO, 2, 36, 36, 3, 3);

  k_flash<<<dim3(81, GG, NB), 32, 0, stream>>>(qT, kTb, vvb, wsScale, aT);

  k_combine<<<dim3(81, GG, NB), 192, 0, stream>>>(aT, waI, waG, waO,
      preI, preG, preO, gain_i, bias_i, gain_g, bias_g, gain_o, bias_o);

  k_final<<<(NB * CC * HWSZ) / 256, 256, 0, stream>>>(preI, preG, preO, (float*)d_out);
}
